// SceneGraphGATLayer_69166153335439
// MI455X (gfx1250) — compile-verified, matrix operand order checked by reading, not run on hardware
//
#include <hip/hip_runtime.h>
#include <math.h>

#define B_   32
#define N_   256
#define D_   512
#define H_   8
#define HD_  64
#define E_   1024
#define EDGE_DIM_ 256
#define GEO_DIM_  64
#define DFF_ 2048
#define M_   (B_ * N_)   // 8192 token rows

typedef unsigned short u16;
typedef unsigned int   u32;
typedef __bf16 bf16;
typedef __attribute__((ext_vector_type(16))) __bf16 v16bf;
typedef __attribute__((ext_vector_type(8)))  float  v8f;
typedef __attribute__((ext_vector_type(4)))  unsigned int u32x4;

// ---------- helpers ----------

__device__ __forceinline__ u16 f2bf(float x) {
  u32 u = __float_as_uint(x);
  u32 r = u + 0x7fffu + ((u >> 16) & 1u);   // round-to-nearest-even
  return (u16)(r >> 16);
}

__device__ __forceinline__ float wave_sum(float v) {
  for (int off = 16; off > 0; off >>= 1) v += __shfl_xor(v, off, 32);
  return v;
}
__device__ __forceinline__ float wave_max(float v) {
  for (int off = 16; off > 0; off >>= 1) {
    float o = __shfl_xor(v, off, 32);
    v = v > o ? v : o;
  }
  return v;
}

union FragU { v16bf v; u32x4 q[2]; };

// Load a 16x32 bf16 fragment (A-layout; identical pattern serves B since
// B[k][n] = W[n][k] and W is row-major). lane l -> row l&15; two contiguous
// 16B chunks at K offsets (l>>4)*8 and 16+(l>>4)*8.
__device__ __forceinline__ v16bf load_frag(const u16* base, int ld) {
  int lane = threadIdx.x & 31;
  int r = lane & 15, hi = lane >> 4;
  const u16* p = base + (size_t)r * ld + hi * 8;
  FragU f;
  f.q[0] = *(const u32x4*)(p);
  f.q[1] = *(const u32x4*)(p + 16);
  return f.v;
}

// async 16B global -> LDS copy (per-lane addresses), tracked by ASYNCcnt
__device__ __forceinline__ void async_g2l_b128(const u16* gptr, const u16* lptr) {
  unsigned lds = (unsigned)(size_t)lptr;                  // LDS aperture: low 32 bits
  unsigned long long ga = (unsigned long long)(size_t)gptr;
  asm volatile("global_load_async_to_lds_b128 %0, %1, off"
               :: "v"(lds), "v"(ga) : "memory");
}
__device__ __forceinline__ void wait_asynccnt0() {
  asm volatile("s_wait_asynccnt 0x0" ::: "memory");
}

// ---------- elementwise / LN kernels ----------

__global__ void f2bf_kernel(const float* __restrict__ in, u16* __restrict__ out, int n) {
  int i = blockIdx.x * blockDim.x + threadIdx.x;
  int stride = gridDim.x * blockDim.x;
  for (; i < n; i += stride) out[i] = f2bf(in[i]);
}

// one wave per row; writes bf16 normalized row
__global__ void ln_rows_kernel(const float* __restrict__ in, const float* __restrict__ g,
                               const float* __restrict__ bta, u16* __restrict__ out,
                               int rows, int cols) {
  int wave = threadIdx.x >> 5;
  int lane = threadIdx.x & 31;
  int row = blockIdx.x * (blockDim.x >> 5) + wave;
  if (row >= rows) return;
  const float* p = in + (size_t)row * cols;
  float s = 0.f, sq = 0.f;
  for (int c = lane; c < cols; c += 32) { float x = p[c]; s += x; sq += x * x; }
  s = wave_sum(s); sq = wave_sum(sq);
  float mean = s / cols;
  float var  = sq / cols - mean * mean;
  float rs   = rsqrtf(var + 1e-5f);
  u16* q = out + (size_t)row * cols;
  for (int c = lane; c < cols; c += 32)
    q[c] = f2bf((p[c] - mean) * rs * g[c] + bta[c]);
}

// ---------- generic WMMA bf16 GEMM: Y = act(X @ W^T + bias) (+res) (*rowmask) ----------
// X: M x K bf16 row-major. W: Nout x K bf16 row-major (original weight layout).
// grid: (Nout/128, M/32), block 256 (8 waves). Each wave owns one 16-wide N tile
// and TWO 16-row M tiles, so every weight (B) fragment feeds two WMMAs.
// A panel (32 x 512) staged in LDS via GLOBAL_LOAD_ASYNC_TO_LDS_B128.

#define KC 512

__global__ void gemm_bf16_kernel(const u16* __restrict__ X, const u16* __restrict__ W,
                                 const float* __restrict__ bias,
                                 const float* __restrict__ residual,
                                 const float* __restrict__ rowmask,
                                 float* __restrict__ outF, u16* __restrict__ outB,
                                 int M, int K, int Nout, int act) {
  __shared__ __align__(16) u16 As[32 * KC];   // 32KB A-panel stage
  int wv   = threadIdx.x >> 5;
  int lane = threadIdx.x & 31;
  int m0    = blockIdx.y * 32;
  int ntile = blockIdx.x * 128 + wv * 16;
  v8f acc0 = {};
  v8f acc1 = {};
  for (int k0 = 0; k0 < K; k0 += KC) {
    // async stage of 32 x KC A panel (KC/8 == 64 16B vectors per row)
    for (int t = threadIdx.x; t < 32 * (KC / 8); t += blockDim.x) {
      int row = t >> 6;
      int c8  = t & 63;
      async_g2l_b128(&X[(size_t)(m0 + row) * K + k0 + c8 * 8],
                     &As[row * KC + c8 * 8]);
    }
    wait_asynccnt0();
    __syncthreads();
    const u16* wbase = W + (size_t)ntile * K + k0;
    __builtin_prefetch(wbase + KC, 0, 0);      // next weight chunk -> L2
    for (int kt = 0; kt < KC; kt += 32) {
      v16bf b  = load_frag(wbase + kt, K);
      v16bf a0 = load_frag(As + kt, KC);
      v16bf a1 = load_frag(As + 16 * KC + kt, KC);
      acc0 = __builtin_amdgcn_wmma_f32_16x16x32_bf16(false, a0, false, b,
                                                     (short)0, acc0, false, false);
      acc1 = __builtin_amdgcn_wmma_f32_16x16x32_bf16(false, a1, false, b,
                                                     (short)0, acc1, false, false);
    }
    __syncthreads();
  }
  int col = ntile + (lane & 15);
  int mr0 = m0 + (lane >> 4) * 8;
  for (int half = 0; half < 2; ++half) {
    v8f acc = half ? acc1 : acc0;
    for (int r = 0; r < 8; ++r) {
      int m = mr0 + half * 16 + r;
      float y = acc[r];
      if (bias) y += bias[col];
      if (act == 1) y = y / (1.f + __expf(-y));       // SiLU
      if (residual) y += residual[(size_t)m * Nout + col];
      if (rowmask)  y *= rowmask[m];
      size_t oi = (size_t)m * Nout + col;
      if (outF) outF[oi] = y;
      if (outB) outB[oi] = f2bf(y);
    }
  }
}

// ---------- edge bias: tb[b,e,h] = emb@W_eb^T + LN(geo)@W_geo^T + biases ----------
// one wave per (b,e)

__global__ void edge_bias_kernel(const float* __restrict__ emb, const float* __restrict__ geo,
                                 const float* __restrict__ gg, const float* __restrict__ gb,
                                 const float* __restrict__ Web, const float* __restrict__ beb,
                                 const float* __restrict__ Wgeo, const float* __restrict__ bgeo,
                                 float* __restrict__ tb) {
  int wave = threadIdx.x >> 5, lane = threadIdx.x & 31;
  int idx = blockIdx.x * 8 + wave;              // b*E + e
  if (idx >= B_ * E_) return;
  const float* gp = geo + (size_t)idx * GEO_DIM_;
  float g0 = gp[lane], g1 = gp[lane + 32];
  float s  = wave_sum(g0 + g1);
  float sq = wave_sum(g0 * g0 + g1 * g1);
  float mean = s / GEO_DIM_;
  float var  = sq / GEO_DIM_ - mean * mean;
  float rs   = rsqrtf(var + 1e-5f);
  float gn0 = (g0 - mean) * rs * gg[lane]      + gb[lane];
  float gn1 = (g1 - mean) * rs * gg[lane + 32] + gb[lane + 32];
  const float* ep = emb + (size_t)idx * EDGE_DIM_;
  for (int h = 0; h < H_; ++h) {
    float acc = 0.f;
    const float* wr = Web + h * EDGE_DIM_;
    for (int c = lane; c < EDGE_DIM_; c += 32) acc += ep[c] * wr[c];
    acc += gn0 * Wgeo[h * GEO_DIM_ + lane] + gn1 * Wgeo[h * GEO_DIM_ + lane + 32];
    acc = wave_sum(acc);
    if (lane == 0) tb[(size_t)idx * H_ + h] = acc + beb[h] + bgeo[h];
  }
}

// ---------- additive bias/mask matrix: -1e30 everywhere, 0 on unmasked diagonal ----------

__global__ void bias_init_kernel(float* __restrict__ bias, const float* __restrict__ nm) {
  size_t t = (size_t)blockIdx.x * blockDim.x + threadIdx.x;
  size_t stride = (size_t)gridDim.x * blockDim.x;
  size_t total = (size_t)B_ * H_ * N_ * N_;
  for (; t < total; t += stride) {
    int j = (int)(t & (N_ - 1));
    size_t r = t >> 8;
    int i = (int)(r & (N_ - 1)); r >>= 8;
    int b = (int)(r >> 3);
    float v = -1e30f;
    if (i == j && nm[b * N_ + i] != 0.f) v = 0.f;
    bias[t] = v;
  }
}

__global__ void bias_scatter_kernel(const int* __restrict__ edges, const float* __restrict__ rmask,
                                    const float* __restrict__ nm, const float* __restrict__ tb,
                                    float* __restrict__ bias) {
  int t = blockIdx.x * blockDim.x + threadIdx.x;
  if (t >= B_ * E_ * H_) return;
  int h = t & (H_ - 1);
  int e = (t >> 3) & (E_ - 1);
  int b = t >> 13;
  int be = b * E_ + e;
  if (rmask[be] == 0.f) return;
  int s = edges[2 * be], d = edges[2 * be + 1];
  s = s < 0 ? 0 : (s > N_ - 1 ? N_ - 1 : s);
  d = d < 0 ? 0 : (d > N_ - 1 ? N_ - 1 : d);
  if (nm[b * N_ + s] == 0.f || nm[b * N_ + d] == 0.f) return;
  float v = tb[(size_t)be * H_ + h];
  size_t base = (size_t)(b * H_ + h) * N_ * N_;
  bias[base + (size_t)s * N_ + d] = v;
  bias[base + (size_t)d * N_ + s] = v;
}

// ---------- fused attention per (b,h): S = QK^T/8 + bias, softmax, O = P V ----------
// block 256 threads; loops over 16 i-tiles; V kept transposed in LDS so the
// PV matmul's B-fragment loads are contiguous.

__global__ void attention_kernel(const u16* __restrict__ qb, const u16* __restrict__ kb,
                                 const u16* __restrict__ vb, const float* __restrict__ bias,
                                 u16* __restrict__ outb) {
  __shared__ __align__(16) u16   VT[HD_][N_];   // 32KB V^T
  __shared__ __align__(16) u16   Qs[16][HD_];   // 2KB
  __shared__ __align__(16) float Ss[16][N_];    // 16KB scores
  __shared__ __align__(16) u16   Ps[16][N_];    // 8KB probs (bf16)
  int b = blockIdx.x >> 3;
  int h = blockIdx.x & 7;
  int wv = threadIdx.x >> 5, lane = threadIdx.x & 31;

  for (int t = threadIdx.x; t < N_ * HD_; t += blockDim.x) {
    int j = t >> 6, d = t & 63;
    VT[d][j] = vb[((size_t)(b * N_ + j)) * D_ + h * HD_ + d];
  }
  __syncthreads();

  const float scale = 0.125f;   // 1/sqrt(64)
  for (int it = 0; it < N_ / 16; ++it) {
    int i0 = it * 16;
    for (int t = threadIdx.x; t < 16 * HD_; t += blockDim.x) {
      int ri = t >> 6, d = t & 63;
      Qs[ri][d] = qb[((size_t)(b * N_ + i0 + ri)) * D_ + h * HD_ + d];
    }
    __syncthreads();

    // scores: each wave does two 16x16 j-tiles
    for (int jj = 0; jj < 2; ++jj) {
      int jt = wv * 2 + jj;
      v8f acc = {};
      const u16* kbase = kb + ((size_t)(b * N_ + jt * 16)) * D_ + h * HD_;
      for (int kt = 0; kt < HD_; kt += 32) {
        v16bf a  = load_frag(&Qs[0][0] + kt, HD_);
        v16bf bf = load_frag(kbase + kt, D_);
        acc = __builtin_amdgcn_wmma_f32_16x16x32_bf16(false, a, false, bf,
                                                      (short)0, acc, false, false);
      }
      int col = jt * 16 + (lane & 15);
      int rb0 = (lane >> 4) * 8;
      const float* bp = bias + ((size_t)(b * H_ + h) * N_ + i0) * N_;
      for (int r = 0; r < 8; ++r)
        Ss[rb0 + r][col] = acc[r] * scale + bp[(size_t)(rb0 + r) * N_ + col];
    }
    __syncthreads();

    // softmax: one wave per 2 rows
    for (int rr = 0; rr < 2; ++rr) {
      int row = wv * 2 + rr;
      float mx = -3.4e38f;
      for (int c = lane; c < N_; c += 32) { float v = Ss[row][c]; mx = mx > v ? mx : v; }
      mx = wave_max(mx);
      float sm = 0.f;
      for (int c = lane; c < N_; c += 32) {
        float p = __expf(Ss[row][c] - mx);
        Ss[row][c] = p;
        sm += p;
      }
      sm = wave_sum(sm);
      float inv = sm > 0.f ? 1.f / sm : 0.f;
      for (int c = lane; c < N_; c += 32) Ps[row][c] = f2bf(Ss[row][c] * inv);
    }
    __syncthreads();

    // O = P @ V : waves 0..3 each own one 16-wide d-tile, K = 256
    if (wv < 4) {
      v8f acc = {};
      for (int kt = 0; kt < N_; kt += 32) {
        v16bf a  = load_frag(&Ps[0][0] + kt, N_);
        v16bf bf = load_frag(&VT[wv * 16][0] + kt, N_);
        acc = __builtin_amdgcn_wmma_f32_16x16x32_bf16(false, a, false, bf,
                                                      (short)0, acc, false, false);
      }
      int dcol = h * HD_ + wv * 16 + (lane & 15);
      int rb0  = (lane >> 4) * 8;
      for (int r = 0; r < 8; ++r)
        outb[((size_t)(b * N_ + i0 + rb0 + r)) * D_ + dcol] = f2bf(acc[r]);
    }
    __syncthreads();
  }
}

// ---------- launch ----------

extern "C" void kernel_launch(void* const* d_in, const int* in_sizes, int n_in,
                              void* d_out, int out_size, void* d_ws, size_t ws_size,
                              hipStream_t stream) {
  (void)in_sizes; (void)n_in; (void)out_size; (void)ws_size;
  const float* x     = (const float*)d_in[0];
  const float* nm    = (const float*)d_in[1];
  const int*   edges = (const int*)  d_in[2];
  const float* rmask = (const float*)d_in[3];
  const float* remb  = (const float*)d_in[4];
  const float* rgeo  = (const float*)d_in[5];
  const float* Wq    = (const float*)d_in[6];  const float* bq   = (const float*)d_in[7];
  const float* Wk    = (const float*)d_in[8];  const float* bk   = (const float*)d_in[9];
  const float* Wv    = (const float*)d_in[10]; const float* bv   = (const float*)d_in[11];
  const float* Web   = (const float*)d_in[12]; const float* beb  = (const float*)d_in[13];
  const float* gg    = (const float*)d_in[14]; const float* gbt  = (const float*)d_in[15];
  const float* Wgeo  = (const float*)d_in[16]; const float* bgeo = (const float*)d_in[17];
  const float* Wout  = (const float*)d_in[18]; const float* bout = (const float*)d_in[19];
  const float* ng    = (const float*)d_in[20]; const float* nb   = (const float*)d_in[21];
  const float* fg    = (const float*)d_in[22]; const float* fb   = (const float*)d_in[23];
  const float* Wff1  = (const float*)d_in[24]; const float* bff1 = (const float*)d_in[25];
  const float* Wff2  = (const float*)d_in[26]; const float* bff2 = (const float*)d_in[27];
  float* out = (float*)d_out;

  char* ws = (char*)d_ws;
  size_t off = 0;
  auto alloc = [&](size_t bytes) -> void* {
    void* p = ws + off;
    off += (bytes + 255) & ~(size_t)255;
    return p;
  };

  u16*   xn   = (u16*)  alloc((size_t)M_ * D_ * 2);
  u16*   wqb  = (u16*)  alloc((size_t)D_ * D_ * 2);
  u16*   wkb  = (u16*)  alloc((size_t)D_ * D_ * 2);
  u16*   wvb  = (u16*)  alloc((size_t)D_ * D_ * 2);
  u16*   wob  = (u16*)  alloc((size_t)D_ * D_ * 2);
  u16*   wf1b = (u16*)  alloc((size_t)DFF_ * D_ * 2);
  u16*   wf2b = (u16*)  alloc((size_t)D_ * DFF_ * 2);
  u16*   qb   = (u16*)  alloc((size_t)M_ * D_ * 2);
  u16*   kbf  = (u16*)  alloc((size_t)M_ * D_ * 2);
  u16*   vbf  = (u16*)  alloc((size_t)M_ * D_ * 2);
  float* tb   = (float*)alloc((size_t)B_ * E_ * H_ * 4);
  float* biasM= (float*)alloc((size_t)B_ * H_ * N_ * N_ * 4);
  u16*   aob  = (u16*)  alloc((size_t)M_ * D_ * 2);
  float* x2   = (float*)alloc((size_t)M_ * D_ * 4);
  u16*   x2n  = (u16*)  alloc((size_t)M_ * D_ * 2);
  u16*   h1   = (u16*)  alloc((size_t)M_ * DFF_ * 2);

  // 1) weights -> bf16
  f2bf_kernel<<<1024, 256, 0, stream>>>(Wq,   wqb,  D_ * D_);
  f2bf_kernel<<<1024, 256, 0, stream>>>(Wk,   wkb,  D_ * D_);
  f2bf_kernel<<<1024, 256, 0, stream>>>(Wv,   wvb,  D_ * D_);
  f2bf_kernel<<<1024, 256, 0, stream>>>(Wout, wob,  D_ * D_);
  f2bf_kernel<<<2048, 256, 0, stream>>>(Wff1, wf1b, DFF_ * D_);
  f2bf_kernel<<<2048, 256, 0, stream>>>(Wff2, wf2b, D_ * DFF_);

  // 2) xn = LN(x)
  ln_rows_kernel<<<M_ / 8, 256, 0, stream>>>(x, ng, nb, xn, M_, D_);

  // 3) q,k,v projections (WMMA)
  dim3 gqkv(D_ / 128, M_ / 32);
  gemm_bf16_kernel<<<gqkv, 256, 0, stream>>>(xn, wqb, bq, nullptr, nullptr, nullptr, qb,  M_, D_, D_, 0);
  gemm_bf16_kernel<<<gqkv, 256, 0, stream>>>(xn, wkb, bk, nullptr, nullptr, nullptr, kbf, M_, D_, D_, 0);
  gemm_bf16_kernel<<<gqkv, 256, 0, stream>>>(xn, wvb, bv, nullptr, nullptr, nullptr, vbf, M_, D_, D_, 0);

  // 4) edge bias values
  edge_bias_kernel<<<(B_ * E_) / 8, 256, 0, stream>>>(remb, rgeo, gg, gbt, Web, beb, Wgeo, bgeo, tb);

  // 5) additive bias/mask matrix
  bias_init_kernel<<<16384, 256, 0, stream>>>(biasM, nm);
  bias_scatter_kernel<<<(B_ * E_ * H_) / 256, 256, 0, stream>>>(edges, rmask, nm, tb, biasM);

  // 6) fused attention
  attention_kernel<<<B_ * H_, 256, 0, stream>>>(qb, kbf, vbf, biasM, aob);

  // 7) x2 = x + attn_out @ Wout^T + bout
  gemm_bf16_kernel<<<gqkv, 256, 0, stream>>>(aob, wob, bout, x, nullptr, x2, nullptr, M_, D_, D_, 0);

  // 8) FF block
  ln_rows_kernel<<<M_ / 8, 256, 0, stream>>>(x2, fg, fb, x2n, M_, D_);
  dim3 gff1(DFF_ / 128, M_ / 32);
  gemm_bf16_kernel<<<gff1, 256, 0, stream>>>(x2n, wf1b, bff1, nullptr, nullptr, nullptr, h1, M_, D_, DFF_, 1);
  gemm_bf16_kernel<<<gqkv, 256, 0, stream>>>(h1, wf2b, bff2, x2, nm, out, nullptr, M_, DFF_, D_, 0);
}
